// ImportanceRenderer_76227079569783
// MI455X (gfx1250) — matrix operand order repaired and checked by reading, not compiled, not run on hardware
//
#include <hip/hip_runtime.h>
#include <hip/hip_bf16.h>

typedef __attribute__((ext_vector_type(16))) _Float16 v16h;
typedef __attribute__((ext_vector_type(8)))  _Float16 v8h;
typedef __attribute__((ext_vector_type(8)))  float    v8f;

#define NB   2
#define NPL  3
#define CCH  32
#define HH   256
#define WW   256
#define RR   16384
#define SS   32
#define PIX  (HH*WW)          /* 65536 */
#define RAY_START 2.25f
#define RAY_END   3.3f

// workspace layout (bytes)
#define PLANES_ELEMS  ((long)NB*NPL*PIX*CCH)        /* 12,582,912 f32 */
#define W1P_OFF       (PLANES_ELEMS*4)              /* 50,331,648 */
#define W1P_ELEMS     (4*32*16)                     /* 2048 halves  */
#define W2P_OFF       (W1P_OFF + W1P_ELEMS*2)
#define W2P_ELEMS     (2*3*32*16)                   /* 3072 halves  */

// ---------------------------------------------------------------------------
// Kernel 1: tri-plane CHW -> HWC so each bilinear corner is one 128B line.
// Reads coalesced across threads per channel; dataset then lives in L2.
// ---------------------------------------------------------------------------
__global__ void k_chw_to_hwc(const float* __restrict__ src, float* __restrict__ dst) {
    int pid   = blockIdx.x * blockDim.x + threadIdx.x;   // 0 .. NB*NPL*PIX-1
    int plane = pid >> 16;
    int pix   = pid & (PIX - 1);
    const float* s = src + (long)plane * CCH * PIX + pix;
    float4 buf[8];
#pragma unroll
    for (int q = 0; q < 8; q++) {
        buf[q].x = s[(q*4 + 0) * PIX];
        buf[q].y = s[(q*4 + 1) * PIX];
        buf[q].z = s[(q*4 + 2) * PIX];
        buf[q].w = s[(q*4 + 3) * PIX];
    }
    float4* d4 = (float4*)(dst + (long)pid * CCH);
#pragma unroll
    for (int q = 0; q < 8; q++) d4[q] = buf[q];
}

// ---------------------------------------------------------------------------
// Kernel 2: pre-swizzle w1[32][64] and w2[64][33] (zero-padded to 48 cols)
// into the wave32 WMMA B-operand register order:
//   b[lane][i] = W[K][N],  N = lane&15,  K = 8*(lane>>4) + (i&7) + 16*(i>>3)
// so each lane loads its v16h operand with one contiguous 32B read.
// ---------------------------------------------------------------------------
__global__ void k_pack(const float* __restrict__ w1, const float* __restrict__ w2,
                       _Float16* __restrict__ w1p, _Float16* __restrict__ w2p) {
    int t = blockIdx.x * blockDim.x + threadIdx.x;
    if (t < W1P_ELEMS) {                       // [j:4][lane:32][i:16]
        int i = t & 15, lane = (t >> 4) & 31, j = t >> 9;
        int n = j * 16 + (lane & 15);
        int k = ((lane >> 4) << 3) + (i & 7) + ((i >> 3) << 4);
        w1p[t] = (_Float16)w1[k * 64 + n];
    }
    if (t < W2P_ELEMS) {                       // [kc:2][j:3][lane:32][i:16]
        int i = t & 15, lane = (t >> 4) & 31;
        int jc = t >> 9;
        int j = jc % 3, kc = jc / 3;
        int n = j * 16 + (lane & 15);
        int k = kc * 32 + (((lane >> 4) << 3) + (i & 7) + ((i >> 3) << 4));
        w2p[t] = (_Float16)((n < 33) ? w2[k * 33 + n] : 0.0f);
    }
}

// ---------------------------------------------------------------------------
// Kernel 3: fused sample -> WMMA MLP -> ray-march.
// Block = 256 thr = 8 waves = 128 samples = 4 rays.
// ---------------------------------------------------------------------------
__device__ __forceinline__ void acc_corner(float acc[16], float w, const float* p) {
#pragma unroll
    for (int q = 0; q < 4; q++) {
        float4 v = ((const float4*)p)[q];
        acc[q*4+0] += w * v.x;  acc[q*4+1] += w * v.y;
        acc[q*4+2] += w * v.z;  acc[q*4+3] += w * v.w;
    }
}
__device__ __forceinline__ float softplus_f(float x) {
    return fmaxf(x, 0.0f) + log1pf(__expf(-fabsf(x)));
}
__device__ __forceinline__ int clampi(int v, int lo, int hi) {
    return v < lo ? lo : (v > hi ? hi : v);
}

__launch_bounds__(256)
__global__ void k_render(const float* __restrict__ planes,   // HWC
                         const float* __restrict__ orig,
                         const float* __restrict__ dirs,
                         const _Float16* __restrict__ w1p,
                         const float* __restrict__ b1,
                         const _Float16* __restrict__ w2p,
                         const float* __restrict__ b2,
                         float* __restrict__ out) {
    __shared__ _Float16 feats[128][32];   // aggregated tri-plane features (f16)
    __shared__ _Float16 hbuf [128][64];   // layer-1 activations (f16)
    __shared__ float    rgbbuf[128][32];  // post-sigmoid rgb features
    __shared__ float    sigbuf[128];      // raw sigma

    const int tid  = threadIdx.x;
    const int lane = tid & 31;
    const int wave = tid >> 5;

    // ---------------- Phase A: tri-plane bilinear sampling -----------------
    {
        int  sLoc = tid >> 1;                 // 0..127 (2 threads / sample)
        int  half = tid & 1;                  // channel half: 16 channels each
        long gs   = (long)blockIdx.x * 128 + sLoc;
        int  n    = (int)(gs / ((long)RR * SS));
        int  ray  = (int)((gs / SS) % RR);
        int  s    = (int)(gs % SS);
        float t   = ((float)s + 0.5f) * (1.0f / SS);
        float dpt = RAY_START + (RAY_END - RAY_START) * t;
        const float* o  = orig + ((long)n * RR + ray) * 3;
        const float* dr = dirs + ((long)n * RR + ray) * 3;
        float x = (o[0] + dpt * dr[0]) * 2.0f;   // * 2/BOX_WARP
        float y = (o[1] + dpt * dr[1]) * 2.0f;
        float z = (o[2] + dpt * dr[2]) * 2.0f;
        float us[3] = { x, y, x };               // xy, yz, xz projections
        float vs[3] = { y, z, z };
        float acc[16];
#pragma unroll
        for (int c = 0; c < 16; c++) acc[c] = 0.0f;
#pragma unroll
        for (int p = 0; p < 3; p++) {
            float ix  = (us[p] + 1.0f) * 0.5f * (float)(WW - 1);
            float iy  = (vs[p] + 1.0f) * 0.5f * (float)(HH - 1);
            float ix0 = floorf(ix), iy0 = floorf(iy);
            float ix1 = ix0 + 1.0f, iy1 = iy0 + 1.0f;
            float wnw = (ix1 - ix) * (iy1 - iy);
            float wne = (ix - ix0) * (iy1 - iy);
            float wsw = (ix1 - ix) * (iy - iy0);
            float wse = (ix - ix0) * (iy - iy0);
            int ix0c = clampi((int)ix0, 0, WW - 1);
            int ix1c = clampi((int)ix1, 0, WW - 1);
            int iy0c = clampi((int)iy0, 0, HH - 1);
            int iy1c = clampi((int)iy1, 0, HH - 1);
            const float* pp = planes + ((long)(n * NPL + p) * PIX) * CCH + half * 16;
            acc_corner(acc, wnw, pp + ((long)(iy0c * WW + ix0c)) * CCH);
            acc_corner(acc, wne, pp + ((long)(iy0c * WW + ix1c)) * CCH);
            acc_corner(acc, wsw, pp + ((long)(iy1c * WW + ix0c)) * CCH);
            acc_corner(acc, wse, pp + ((long)(iy1c * WW + ix1c)) * CCH);
        }
#pragma unroll
        for (int c = 0; c < 16; c++)
            feats[sLoc][half * 16 + c] = (_Float16)(acc[c] * (1.0f / 3.0f));
    }
    __syncthreads();

    // ---------------- Phase B: MLP via v_wmma_f32_16x16x32_f16 -------------
    // Each wave owns 16 samples (rows wave*16 .. wave*16+15).
    // A layout (16-bit, 16x32): lane<16 -> M=lane, K={0..7,16..23};
    //                           lane>=16 -> M=lane-16, K={8..15,24..31}.
    // C/D layout (f32 16x16):   vgpr r -> M = r + 8*(lane>=16), N = lane&15.
    const int M   = lane & 15;
    const int kb0 = (lane < 16) ? 0 : 8;
    const int mhi = (lane >> 4) << 3;

    v16h a1;
    {
        v8h lo = *(const v8h*)&feats[wave * 16 + M][kb0];
        v8h hi = *(const v8h*)&feats[wave * 16 + M][kb0 + 16];
#pragma unroll
        for (int i = 0; i < 8; i++) { a1[i] = lo[i]; a1[8 + i] = hi[i]; }
    }
#pragma unroll
    for (int j = 0; j < 4; j++) {                       // layer 1: N=64 in 4 tiles
        v16h b = *(const v16h*)(w1p + (j * 32 + lane) * 16);
        v8f  c = {};
        c = __builtin_amdgcn_wmma_f32_16x16x32_f16(false, a1, false, b,
                                                   (short)0, c, false, false);
        float bias = b1[j * 16 + M];
#pragma unroll
        for (int r = 0; r < 8; r++) {
            float h = softplus_f(c[r] + bias);
            hbuf[wave * 16 + r + mhi][j * 16 + M] = (_Float16)h;
        }
    }
    __syncthreads();

    v8f oacc[3] = { {}, {}, {} };
#pragma unroll
    for (int kc = 0; kc < 2; kc++) {                    // layer 2: K=64 in 2 chunks
        v16h a2;
        {
            v8h lo = *(const v8h*)&hbuf[wave * 16 + M][kc * 32 + kb0];
            v8h hi = *(const v8h*)&hbuf[wave * 16 + M][kc * 32 + kb0 + 16];
#pragma unroll
            for (int i = 0; i < 8; i++) { a2[i] = lo[i]; a2[8 + i] = hi[i]; }
        }
#pragma unroll
        for (int j = 0; j < 3; j++) {                   // N=48 (33 valid) in 3 tiles
            v16h b = *(const v16h*)(w2p + ((kc * 3 + j) * 32 + lane) * 16);
            oacc[j] = __builtin_amdgcn_wmma_f32_16x16x32_f16(false, a2, false, b,
                                                (short)0, oacc[j], false, false);
        }
    }
    // epilogue: bias + sigma/rgb split, write to LDS for the ray-marcher
#pragma unroll
    for (int j = 0; j < 3; j++) {
        int gcol   = j * 16 + M;
        float bias = (gcol < 33) ? b2[gcol] : 0.0f;
#pragma unroll
        for (int r = 0; r < 8; r++) {
            int row  = wave * 16 + r + mhi;
            float vv = oacc[j][r] + bias;
            if (gcol == 0) {
                sigbuf[row] = vv;
            } else if (gcol < 33) {
                float sg = 1.0f / (1.0f + __expf(-vv));
                rgbbuf[row][gcol - 1] = sg * 1.002f - 0.001f;
            }
        }
    }
    __syncthreads();

    // ---------------- Phase C: alpha compositing (MipRayMarcher2) ----------
    if (wave < 4) {                       // 4 rays per block; lane = channel
        int  rLoc = wave;
        long gs0  = (long)blockIdx.x * 128 + rLoc * 32;
        int  n    = (int)(gs0 / ((long)RR * SS));
        int  ray  = (int)((gs0 / SS) % RR);
        const float delta = (RAY_END - RAY_START) * (1.0f / SS);
        float T = 1.0f, acc = 0.0f;
#pragma unroll 1
        for (int s = 0; s < SS - 1; s++) {
            float sgm   = 0.5f * (sigbuf[rLoc*32 + s] + sigbuf[rLoc*32 + s + 1]) - 1.0f;
            float dens  = softplus_f(sgm);
            float alpha = 1.0f - __expf(-delta * dens);
            float cmid  = 0.5f * (rgbbuf[rLoc*32 + s][lane] + rgbbuf[rLoc*32 + s + 1][lane]);
            acc += alpha * T * cmid;
            T   *= (1.0f - alpha + 1e-10f);
        }
        out[((long)n * RR + ray) * CCH + lane] = acc;
    }
}

// ---------------------------------------------------------------------------
extern "C" void kernel_launch(void* const* d_in, const int* in_sizes, int n_in,
                              void* d_out, int out_size, void* d_ws, size_t ws_size,
                              hipStream_t stream) {
    const float* vol  = (const float*)d_in[0];   // [2,3,32,256,256]
    const float* orig = (const float*)d_in[1];   // [2,16384,3]
    const float* dirs = (const float*)d_in[2];   // [2,16384,3]
    const float* w1   = (const float*)d_in[3];   // [32,64]
    const float* b1   = (const float*)d_in[4];   // [64]
    const float* w2   = (const float*)d_in[5];   // [64,33]
    const float* b2   = (const float*)d_in[6];   // [33]
    float* out = (float*)d_out;

    char* ws = (char*)d_ws;
    float*    planesHWC = (float*)ws;
    _Float16* w1p       = (_Float16*)(ws + W1P_OFF);
    _Float16* w2p       = (_Float16*)(ws + W2P_OFF);

    k_chw_to_hwc<<<(NB * NPL * PIX) / 256, 256, 0, stream>>>(vol, planesHWC);
    k_pack<<<12, 256, 0, stream>>>(w1, w2, w1p, w2p);
    k_render<<<((long)NB * RR * SS) / 128, 256, 0, stream>>>(
        planesHWC, orig, dirs, w1p, b1, w2p, b2, out);
    (void)in_sizes; (void)n_in; (void)out_size; (void)ws_size;
}
